// FeatureExtract_26233660244203
// MI455X (gfx1250) — matrix operand order repaired and checked by reading, not compiled, not run on hardware
//
#include <hip/hip_runtime.h>
#include <math.h>

// ---------------------------------------------------------------------------
// GNN layer on MI455X (gfx1250, wave32, WMMA).
// Key algebraic hoist: msg = gelu(h[src]@W1+b1) depends only on src, so the
// edge-GEMM (42 GFLOP) becomes a node-GEMM (2.6 GFLOP) + memory-bound scatter.
// GEMMs use v_wmma_f32_16x16x32_f16 (f16 in, f32 accumulate).
// ---------------------------------------------------------------------------

typedef __attribute__((ext_vector_type(16))) _Float16 v16h;
typedef __attribute__((ext_vector_type(8)))  _Float16 v8h;
typedef __attribute__((ext_vector_type(8)))  float    v8f;

#define N_NODES  20000
#define N_EDGES  320000
#define CH       256
#define FEAT     118
#define FEAT_PAD 128

__device__ __forceinline__ float gelu_exact(float v) {
    return 0.5f * v * (1.0f + erff(v * 0.70710678118654752f));
}

// ---- convert + zero-pad x: fp32 [N,118] -> f16 [N,128] --------------------
__global__ void cvt_x_kernel(const float* __restrict__ x, _Float16* __restrict__ x16) {
    int tid = blockIdx.x * blockDim.x + threadIdx.x;
    if (tid >= N_NODES * FEAT_PAD) return;
    int row = tid >> 7, k = tid & (FEAT_PAD - 1);
    x16[tid] = (k < FEAT) ? (_Float16)x[row * FEAT + k] : (_Float16)0.0f;
}

// ---- pack fp32 weight W[K,256] (row-major) into WMMA-B lane layout --------
// 16-bit B 32x16 tile layout (wave32): element j of lane n holds B[K][N] with
// K = 8g+j (j<8) or 16+8g+(j-8) (j>=8), g = lane>>4, N = n&15. Flattened so that
// each lane's 16 halfs for one (n_tile, kc) are contiguous (one 32B load).
__global__ void pack_b_kernel(const float* __restrict__ W, _Float16* __restrict__ Bp,
                              int KC, int Kact) {
    int tid = blockIdx.x * blockDim.x + threadIdx.x;
    if (tid >= 16 * KC * 512) return;
    int j      = tid & 15;
    int lane   = (tid >> 4) & 31;
    int kc     = (tid >> 9) % KC;
    int n_tile = tid / (KC * 512);
    int k = kc * 32 + ((lane >> 4) << 3) + (j & 7) + ((j >> 3) << 4);
    int n = n_tile * 16 + (lane & 15);
    Bp[tid] = (k < Kact) ? (_Float16)W[k * CH + n] : (_Float16)0.0f;
}

// ---- WMMA GEMM: out = act(A[M,K] @ B[K,256] + bias) -----------------------
// One wave computes one 16x16 tile. MODE 0: store f16. MODE 1: gelu, store
// f32. MODE 2: gelu, mean over the 256 channels, atomic into outF[row].
template <int KC, int MODE>
__global__ void gemm_wmma_kernel(const _Float16* __restrict__ A,
                                 const _Float16* __restrict__ Bp,
                                 const float* __restrict__ bias,
                                 _Float16* __restrict__ outH,
                                 float* __restrict__ outF) {
    constexpr int K = KC * 32;
    const int lane   = threadIdx.x & 31;
    const int wg     = blockIdx.x * (blockDim.x >> 5) + (threadIdx.x >> 5);
    const int m_tile = wg >> 4;          // 16 n_tiles cover CH=256
    const int n_tile = wg & 15;
    const int g      = lane >> 4;
    const int row    = m_tile * 16 + (lane & 15);   // A: lane L holds M = L%16

    union { v16h v; v8h h[2]; } a;
    v8f acc = {};
    const _Float16* arow = A + (size_t)row * K;
    const _Float16* bp   = Bp + (size_t)(n_tile * KC) * 512 + lane * 16;
#pragma unroll
    for (int kc = 0; kc < KC; ++kc) {
        const int k0 = kc * 32;
        // 16-bit A 16x32 layout: lane holds K in [k0+8g, k0+8g+8) and
        // [k0+16+8g, k0+16+8g+8)  (two contiguous 16B segments)
        a.h[0] = *(const v8h*)(arow + k0 + 8 * g);
        a.h[1] = *(const v8h*)(arow + k0 + 16 + 8 * g);
        v16h b = *(const v16h*)(bp + kc * 512);
        acc = __builtin_amdgcn_wmma_f32_16x16x32_f16(
            false, a.v, false, b, (short)0, acc, false, false);
    }

    const int col = n_tile * 16 + (lane & 15);
    const float bc = bias[col];
    // C/D layout: VGPR r holds row M = r + 8g, col = lane&15 (per n_tile)
#pragma unroll
    for (int r = 0; r < 8; ++r) {
        float v = acc[r] + bc;
        const int orow = m_tile * 16 + 8 * g + r;
        if (MODE == 0) {
            outH[(size_t)orow * CH + col] = (_Float16)v;
        } else if (MODE == 1) {
            outF[(size_t)orow * CH + col] = gelu_exact(v);
        } else {
            v = gelu_exact(v);
            // sum the 16 columns held by this half-wave (width-16 butterfly)
            v += __shfl_xor(v, 1, 16);
            v += __shfl_xor(v, 2, 16);
            v += __shfl_xor(v, 4, 16);
            v += __shfl_xor(v, 8, 16);
            if ((lane & 15) == 0)
                atomicAdd(&outF[orow], v * (1.0f / (float)CH));
        }
    }
}

// ---- edge scatter: agg[dst] += m[src]; cnt[dst] += 1 ----------------------
// thread = (edge, 4-channel chunk): float4 gather + 4 f32 atomic adds (in L2)
__global__ void scatter_kernel(const int* __restrict__ ei,
                               const float* __restrict__ m,
                               float* __restrict__ agg,
                               float* __restrict__ cnt) {
    int t = blockIdx.x * blockDim.x + threadIdx.x;
    if (t >= N_EDGES * (CH / 4)) return;
    int e = t >> 6, q = t & 63;
    int src = ei[e];
    int dst = ei[N_EDGES + e];
    float4 v = ((const float4*)m)[(size_t)src * (CH / 4) + q];
    float* dp = agg + (size_t)dst * CH + q * 4;
    atomicAdd(dp + 0, v.x);
    atomicAdd(dp + 1, v.y);
    atomicAdd(dp + 2, v.z);
    atomicAdd(dp + 3, v.w);
    if (q == 0) atomicAdd(cnt + dst, 1.0f);
}

// ---- agg / max(cnt,1) -> f16 ----------------------------------------------
__global__ void normalize_kernel(const float* __restrict__ agg,
                                 const float* __restrict__ cnt,
                                 _Float16* __restrict__ a16) {
    int tid = blockIdx.x * blockDim.x + threadIdx.x;
    if (tid >= N_NODES * CH) return;
    float inv = 1.0f / fmaxf(cnt[tid >> 8], 1.0f);
    a16[tid] = (_Float16)(agg[tid] * inv);
}

extern "C" void kernel_launch(void* const* d_in, const int* in_sizes, int n_in,
                              void* d_out, int out_size, void* d_ws, size_t ws_size,
                              hipStream_t stream) {
    const float* x       = (const float*)d_in[0];
    const int*   ei      = (const int*)d_in[1];     // [2, E] (src row, dst row)
    const float* W_embed = (const float*)d_in[2];
    const float* b_embed = (const float*)d_in[3];
    const float* W1      = (const float*)d_in[4];
    const float* b1      = (const float*)d_in[5];
    const float* W2      = (const float*)d_in[6];
    const float* b2      = (const float*)d_in[7];
    float* out = (float*)d_out;

    // workspace carve-out (256B aligned), ~57 MB total
    char* ws = (char*)d_ws;
    auto carve = [&](size_t bytes) {
        char* p = ws;
        ws += (bytes + 255) & ~(size_t)255;
        return p;
    };
    _Float16* x16 = (_Float16*)carve((size_t)N_NODES * FEAT_PAD * 2);
    _Float16* h16 = (_Float16*)carve((size_t)N_NODES * CH * 2);   // reused as a16
    float*    mbu = (float*)   carve((size_t)N_NODES * CH * 4);
    float*    agg = (float*)   carve((size_t)N_NODES * CH * 4);
    float*    cnt = (float*)   carve((size_t)N_NODES * 4);
    _Float16* WeP = (_Float16*)carve((size_t)16 * 4 * 512 * 2);
    _Float16* W1P = (_Float16*)carve((size_t)16 * 8 * 512 * 2);
    _Float16* W2P = (_Float16*)carve((size_t)16 * 8 * 512 * 2);

    hipMemsetAsync(agg, 0, (size_t)N_NODES * CH * 4, stream);
    hipMemsetAsync(cnt, 0, (size_t)N_NODES * 4, stream);
    hipMemsetAsync(out, 0, (size_t)N_NODES * 4, stream);

    cvt_x_kernel<<<(N_NODES * FEAT_PAD + 255) / 256, 256, 0, stream>>>(x, x16);
    pack_b_kernel<<<(16 * 4 * 512 + 255) / 256, 256, 0, stream>>>(W_embed, WeP, 4, FEAT);
    pack_b_kernel<<<(16 * 8 * 512 + 255) / 256, 256, 0, stream>>>(W1, W1P, 8, CH);
    pack_b_kernel<<<(16 * 8 * 512 + 255) / 256, 256, 0, stream>>>(W2, W2P, 8, CH);

    // (N/16 m_tiles) * 16 n_tiles waves, 8 waves per 256-thread block
    const int gemm_blocks = (N_NODES / 16) * 16 / 8;   // 2500

    // h = x @ W_embed + b_embed            -> f16
    gemm_wmma_kernel<4, 0><<<gemm_blocks, 256, 0, stream>>>(x16, WeP, b_embed, h16, nullptr);
    // m = gelu(h @ W1 + b1)                -> f32   (per-NODE, hoisted off edges)
    gemm_wmma_kernel<8, 1><<<gemm_blocks, 256, 0, stream>>>(h16, W1P, b1, nullptr, mbu);
    // agg[dst] += m[src]; cnt[dst] += 1
    scatter_kernel<<<(N_EDGES * (CH / 4) + 255) / 256, 256, 0, stream>>>(ei, mbu, agg, cnt);
    // a = agg / max(cnt,1)                 -> f16 (reuse h16 buffer)
    normalize_kernel<<<(N_NODES * CH + 255) / 256, 256, 0, stream>>>(agg, cnt, h16);
    // out[row] = mean_c gelu(a @ W2 + b2)
    gemm_wmma_kernel<8, 2><<<gemm_blocks, 256, 0, stream>>>(h16, W2P, b2, nullptr, out);
}
